// GlobalExtractor_42082089566724
// MI455X (gfx1250) — compile-verified, matrix operand order checked by reading, not run on hardware
//
#include <hip/hip_runtime.h>
#include <hip/hip_bf16.h>

// ---------------- problem constants ----------------
#define BSZ   2
#define CIN   64
#define HIMG  96
#define LSEQ  (HIMG*HIMG)          // 9216
#define DI    128                  // d_inner
#define HALFC 64                   // mixer half width
#define NST   16                   // d_state
#define DTR   4                    // dt_rank
#define R36   (DTR + 2*NST)        // 36
#define ROWS  (2*BSZ*LSEQ)         // 36864 rows across {s1,s2} x batch x L
#define LN_EPS 1e-5f
#define NCHUNK 32
#define CHLEN  (LSEQ / NCHUNK)     // 288

typedef __attribute__((ext_vector_type(16))) _Float16 v16h;
typedef __attribute__((ext_vector_type(8)))  float    v8f;

// K index pattern for 16-bit A/B fragments (wave32, 16x16x32):
// lanes 0-15 hold K base 0, lanes 16-31 hold K base 8; VGPRs 0-3 cover K 0..7(+base),
// VGPRs 4-7 cover K 16..23(+base); each VGPR packs K and K+1.
__device__ __forceinline__ int kpat(int lane, int v) {
  return ((lane & 16) ? 8 : 0) + ((v & 4) ? 16 : 0) + 2 * (v & 3);
}

__device__ __forceinline__ float softplusf(float s) {
  return (s > 20.f) ? s : log1pf(expf(s));
}

// ---------------------------------------------------------------------------
// Kernel 1: fused in-projection GEMMs.  X[ROWS,64] @ {Wi^T, Wg^T} -> Ym,Yg [ROWS,128]
// X row (src,b,l), col k comes from f_src[b][k][l] (fp32, converted to f16).
// One wave per (16-row tile, 16-col tile); ntile<8 -> Wi, ntile>=8 -> Wg.
// ---------------------------------------------------------------------------
__global__ void inproj_gemm(const float* __restrict__ f1, const float* __restrict__ f2,
                            const float* __restrict__ Wi, const float* __restrict__ bi,
                            const float* __restrict__ Wg, const float* __restrict__ bg,
                            float* __restrict__ Ym, float* __restrict__ Yg) {
  const int rowTile = blockIdx.x;        // 0..ROWS/16-1
  const int nt      = blockIdx.y;        // 0..15
  const int lane    = threadIdx.x;       // 0..31
  const bool isG    = nt >= 8;
  const float* W    = isG ? Wg : Wi;
  const float* bias = isG ? bg : bi;
  float* Y          = isG ? Yg : Ym;
  const int ncol    = (isG ? nt - 8 : nt) * 16 + (lane & 15);

  const int m0  = rowTile * 16;
  const int mA  = m0 + (lane & 15);
  const int src = mA / (BSZ * LSEQ);
  const int rem = mA - src * (BSZ * LSEQ);
  const int b   = rem / LSEQ;
  const int l   = rem - b * LSEQ;
  const float* xrow = (src ? f2 : f1) + (size_t)b * CIN * LSEQ + l;  // elem k at xrow[k*LSEQ]

  v8f c = {};
  #pragma unroll
  for (int kb = 0; kb < CIN; kb += 32) {
    v16h a, bf;
    #pragma unroll
    for (int v = 0; v < 8; ++v) {
      const int k = kb + kpat(lane, v);
      a[2*v]    = (_Float16)xrow[(size_t)k * LSEQ];
      a[2*v+1]  = (_Float16)xrow[(size_t)(k + 1) * LSEQ];
      bf[2*v]   = (_Float16)W[ncol * CIN + k];
      bf[2*v+1] = (_Float16)W[ncol * CIN + k + 1];
    }
    c = __builtin_amdgcn_wmma_f32_16x16x32_f16(false, a, false, bf, (short)0, c, false, false);
  }
  const float bv = bias[ncol];
  #pragma unroll
  for (int r = 0; r < 8; ++r) {
    const int m = m0 + r + ((lane & 16) ? 8 : 0);
    Y[(size_t)m * DI + ncol] = c[r] + bv;
  }
}

// ---------------------------------------------------------------------------
// Kernel 2: LayerNorm (global path) + transpose both paths to [src,b,ch,L].
// ---------------------------------------------------------------------------
__global__ void ln_transpose(const float* __restrict__ Ym, const float* __restrict__ Yg,
                             const float* __restrict__ lng, const float* __restrict__ lnb,
                             float* __restrict__ xzT, float* __restrict__ ghT) {
  const int row = blockIdx.x;           // 0..ROWS-1
  const int ch  = threadIdx.x;          // 0..127
  const int src = row / (BSZ * LSEQ);
  const int rem = row % (BSZ * LSEQ);
  const int b   = rem / LSEQ;
  const int l   = rem % LSEQ;
  const float vm = Ym[(size_t)row * DI + ch];
  const float vg = Yg[(size_t)row * DI + ch];
  __shared__ float s1[DI], s2[DI];
  s1[ch] = vg; s2[ch] = vg * vg;
  __syncthreads();
  for (int off = 64; off > 0; off >>= 1) {
    if (ch < off) { s1[ch] += s1[ch + off]; s2[ch] += s2[ch + off]; }
    __syncthreads();
  }
  const float mu  = s1[0] * (1.0f / DI);
  const float var = s2[0] * (1.0f / DI) - mu * mu;
  const float h = (vg - mu) * rsqrtf(var + LN_EPS) * lng[ch] + lnb[ch];
  const size_t t = ((size_t)(src * BSZ + b) * DI + ch) * LSEQ + l;
  xzT[t] = vm;
  ghT[t] = h;
}

// ---------------------------------------------------------------------------
// Kernel 3: depthwise conv1d K=4 ('same': pad lo=1,hi=2) + SiLU, channel-major.
// ---------------------------------------------------------------------------
__global__ void dwconv_silu(const float* __restrict__ X,
                            const float* __restrict__ w0, const float* __restrict__ b0,
                            const float* __restrict__ w1, const float* __restrict__ b1,
                            float* __restrict__ Yc) {
  const int idx = blockIdx.x * blockDim.x + threadIdx.x;  // < 2*BSZ*DI*LSEQ
  const int l   = idx % LSEQ;
  const int ch  = (idx / LSEQ) % DI;
  const int sb  = idx / (LSEQ * DI);
  const float* w = (ch < HALFC) ? (w0 + ch * 4) : (w1 + (ch - HALFC) * 4);
  float acc = (ch < HALFC) ? b0[ch] : b1[ch - HALFC];
  const float* xr = X + ((size_t)sb * DI + ch) * LSEQ;
  #pragma unroll
  for (int j = 0; j < 4; ++j) {
    const int p = l - 1 + j;
    if (p >= 0 && p < LSEQ) acc += xr[p] * w[j];
  }
  Yc[((size_t)sb * DI + ch) * LSEQ + l] = acc / (1.0f + expf(-acc));   // SiLU
}

// ---------------------------------------------------------------------------
// Kernel 4: x_dbl[row,36] = sum_d conv[sb,d,l] * Wxp[r,d]
// ---------------------------------------------------------------------------
__global__ void xdbl_proj(const float* __restrict__ Xc, const float* __restrict__ Wxp,
                          const int Dp, float* __restrict__ xdbl) {
  const int idx = blockIdx.x * blockDim.x + threadIdx.x;  // < ROWS*36
  const int r   = idx % R36;
  const int row = idx / R36;                               // (sb,l) row
  const int sb  = row / LSEQ;
  const int l   = row % LSEQ;
  const float* x = Xc + (size_t)sb * DI * LSEQ + l;        // channel d at x[d*LSEQ]
  const float* w = Wxp + r * Dp;
  float acc = 0.f;
  for (int d = 0; d < Dp; ++d) acc += x[(size_t)d * LSEQ] * w[d];
  xdbl[(size_t)row * R36 + r] = acc;
}

// ---------------------------------------------------------------------------
// Kernel 5: dt[sb,d,l] = sum_{r<4} x_dbl[row,r] * Wdt[d,r]
// ---------------------------------------------------------------------------
__global__ void dt_proj(const float* __restrict__ xdbl, const float* __restrict__ Wdt,
                        const int Dp, float* __restrict__ dtT) {
  const int idx = blockIdx.x * blockDim.x + threadIdx.x;  // < 2*BSZ*Dp*LSEQ
  const int l   = idx % LSEQ;
  const int d   = (idx / LSEQ) % Dp;
  const int sb  = idx / (LSEQ * Dp);
  const float* xr = xdbl + ((size_t)sb * LSEQ + l) * R36;
  float acc = 0.f;
  #pragma unroll
  for (int r = 0; r < DTR; ++r) acc += xr[r] * Wdt[d * DTR + r];
  dtT[idx] = acc;
}

// ---------------------------------------------------------------------------
// Kernels 6a/6b/6c: chunked parallel selective scan (exact, linear recurrence).
// Lane mapping in every scan kernel: n = lane&15, d = d0 + (lane>>4) (2 channels/wave).
//
// 6a: per (sb, d-pair, chunk): local scan with h0=0 -> chunk end state + decay product.
// 6b: per (sb, d-pair): serial carry combine over the 32 chunks (32 steps).
// 6c: per (sb, d-pair, chunk): re-run recurrence with carry-in, emit y.
// ---------------------------------------------------------------------------
__global__ void scan_phase1(const float* __restrict__ dtT, const float* __restrict__ Xc,
                            const float* __restrict__ xdbl,
                            const float* __restrict__ Alog, const float* __restrict__ dt_bias,
                            const int Dp, float* __restrict__ Hend, float* __restrict__ Pprod) {
  const int lane   = threadIdx.x;
  const int dpairs = Dp >> 1;
  const int sb     = blockIdx.x / dpairs;
  const int d      = (blockIdx.x % dpairs) * 2 + (lane >> 4);
  const int n      = lane & 15;
  const int c      = blockIdx.y;

  const float aco = -expf(Alog[d * NST + n]);
  const float dtb = dt_bias[d];
  const float* dtrow = dtT  + ((size_t)sb * Dp + d) * LSEQ;
  const float* urow  = Xc   + ((size_t)sb * DI + d) * LSEQ;
  const float* xd    = xdbl + (size_t)sb * LSEQ * R36;

  float h = 0.f, P = 1.f;
  const int l0 = c * CHLEN;
  for (int l = l0; l < l0 + CHLEN; ++l) {
    if ((l & 63) == 0 && l + 128 < l0 + CHLEN)
      __builtin_prefetch(dtrow + l + 128, 0, 1);             // global_prefetch_b8
    const float delta = softplusf(dtrow[l] + dtb);
    const float dA    = expf(delta * aco);
    h = dA * h + delta * xd[(size_t)l * R36 + DTR + n] * urow[l];
    P *= dA;
  }
  const size_t idx = (((size_t)sb * Dp + d) * NST + n) * NCHUNK + c;
  Hend[idx]  = h;
  Pprod[idx] = P;
}

__global__ void scan_combine(const float* __restrict__ Hend, const float* __restrict__ Pprod,
                             const int Dp, float* __restrict__ Hin) {
  const int lane   = threadIdx.x;
  const int dpairs = Dp >> 1;
  const int sb     = blockIdx.x / dpairs;
  const int d      = (blockIdx.x % dpairs) * 2 + (lane >> 4);
  const int n      = lane & 15;
  const size_t base = (((size_t)sb * Dp + d) * NST + n) * NCHUNK;
  float carry = 0.f;
  #pragma unroll 4
  for (int c = 0; c < NCHUNK; ++c) {
    Hin[base + c] = carry;
    carry = Pprod[base + c] * carry + Hend[base + c];
  }
}

__global__ void scan_phase2(const float* __restrict__ dtT, const float* __restrict__ Xc,
                            const float* __restrict__ xdbl,
                            const float* __restrict__ Alog, const float* __restrict__ Dvec,
                            const float* __restrict__ dt_bias, const float* __restrict__ Hin,
                            const int Dp, float* __restrict__ Y) {
  const int lane   = threadIdx.x;
  const int dpairs = Dp >> 1;
  const int sb     = blockIdx.x / dpairs;
  const int d      = (blockIdx.x % dpairs) * 2 + (lane >> 4);
  const int n      = lane & 15;
  const int c      = blockIdx.y;

  const float aco = -expf(Alog[d * NST + n]);
  const float Dv  = Dvec[d];
  const float dtb = dt_bias[d];
  const float* dtrow = dtT  + ((size_t)sb * Dp + d) * LSEQ;
  const float* urow  = Xc   + ((size_t)sb * DI + d) * LSEQ;
  const float* xd    = xdbl + (size_t)sb * LSEQ * R36;
  float* yrow        = Y    + ((size_t)sb * Dp + d) * LSEQ;

  float h = Hin[(((size_t)sb * Dp + d) * NST + n) * NCHUNK + c];
  const int l0 = c * CHLEN;
  for (int l = l0; l < l0 + CHLEN; ++l) {
    if ((l & 63) == 0 && l + 128 < l0 + CHLEN)
      __builtin_prefetch(urow + l + 128, 0, 1);              // global_prefetch_b8
    const float delta = softplusf(dtrow[l] + dtb);
    const float u     = urow[l];
    const float dA    = expf(delta * aco);
    h = dA * h + delta * xd[(size_t)l * R36 + DTR + n] * u;
    float p = h * xd[(size_t)l * R36 + DTR + NST + n];
    p += __shfl_xor(p, 1, 16);
    p += __shfl_xor(p, 2, 16);
    p += __shfl_xor(p, 4, 16);
    p += __shfl_xor(p, 8, 16);
    if (n == 0) yrow[l] = p + Dv * u;
  }
}

// ---------------------------------------------------------------------------
// Kernel 7: output GEMM with fused Hadamard product.
// ---------------------------------------------------------------------------
__global__ void outproj_gemm(const float* __restrict__ yM, const float* __restrict__ convM,
                             const float* __restrict__ yG,
                             const float* __restrict__ Wo, const float* __restrict__ bo,
                             float* __restrict__ outp) {
  const int rowTile = blockIdx.x;        // 0..BSZ*LSEQ/16-1
  const int nt      = blockIdx.y;        // 0..3
  const int which   = blockIdx.z;        // 0 -> o1, 1 -> o2
  const int lane    = threadIdx.x;
  const int srcm = which;                // o1: mixer(s1); o2: mixer(s2)
  const int srcg = 1 - which;            // o1: global(s2); o2: global(s1)

  const int m0 = rowTile * 16;
  const int mA = m0 + (lane & 15);
  const int b  = mA / LSEQ;
  const int l  = mA % LSEQ;
  const int ncol = nt * 16 + (lane & 15);

  const float* ymr = yM    + (size_t)(srcm * BSZ + b) * HALFC * LSEQ + l;
  const float* zr  = convM + (size_t)(srcm * BSZ + b) * DI    * LSEQ + l;  // ch 64..127 = z
  const float* ygr = yG    + (size_t)(srcg * BSZ + b) * DI    * LSEQ + l;

  v8f c = {};
  #pragma unroll
  for (int kb = 0; kb < DI; kb += 32) {
    v16h a, bf;
    #pragma unroll
    for (int v = 0; v < 8; ++v) {
      const int k = kb + kpat(lane, v);
      #pragma unroll
      for (int t = 0; t < 2; ++t) {
        const int kk = k + t;
        const float x11 = (kk < HALFC) ? ymr[(size_t)kk * LSEQ] : zr[(size_t)kk * LSEQ];
        const float x12 = ygr[(size_t)kk * LSEQ];
        a[2*v + t]  = (_Float16)(x11 * x12);
        bf[2*v + t] = (_Float16)Wo[ncol * DI + kk];
      }
    }
    c = __builtin_amdgcn_wmma_f32_16x16x32_f16(false, a, false, bf, (short)0, c, false, false);
  }
  const float bv = bo[ncol];
  #pragma unroll
  for (int r = 0; r < 8; ++r) {
    const int m  = m0 + r + ((lane & 16) ? 8 : 0);
    const int bb = m / LSEQ;
    const int ll = m % LSEQ;
    outp[(size_t)which * BSZ * 64 * LSEQ + ((size_t)(bb * 64 + ncol)) * LSEQ + ll] = c[r] + bv;
  }
}

// ---------------------------------------------------------------------------
extern "C" void kernel_launch(void* const* d_in, const int* in_sizes, int n_in,
                              void* d_out, int out_size, void* d_ws, size_t ws_size,
                              hipStream_t stream) {
  (void)in_sizes; (void)n_in; (void)out_size; (void)ws_size;
  const float* f1     = (const float*)d_in[0];
  const float* f2     = (const float*)d_in[1];
  const float* Wi     = (const float*)d_in[2];
  const float* bi     = (const float*)d_in[3];
  const float* wcx    = (const float*)d_in[4];
  const float* bcx    = (const float*)d_in[5];
  const float* wcz    = (const float*)d_in[6];
  const float* bcz    = (const float*)d_in[7];
  const float* Wxp_m  = (const float*)d_in[8];
  const float* Wdt_m  = (const float*)d_in[9];
  const float* bdt_m  = (const float*)d_in[10];
  const float* Alog_m = (const float*)d_in[11];
  const float* Dm     = (const float*)d_in[12];
  const float* Wg     = (const float*)d_in[13];
  const float* bg     = (const float*)d_in[14];
  const float* ln_g   = (const float*)d_in[15];
  const float* ln_b   = (const float*)d_in[16];
  const float* wcg    = (const float*)d_in[17];
  const float* bcg    = (const float*)d_in[18];
  const float* Wxp_g  = (const float*)d_in[19];
  const float* Wdt_g  = (const float*)d_in[20];
  const float* bdt_g  = (const float*)d_in[21];
  const float* Alog_g = (const float*)d_in[22];
  const float* Dg     = (const float*)d_in[23];
  const float* Wo     = (const float*)d_in[24];
  const float* bo     = (const float*)d_in[25];
  float* out = (float*)d_out;
  float* ws  = (float*)d_ws;

  // workspace layout (floats); buffers reused once their producers/consumers retire
  const size_t RD   = (size_t)ROWS * DI;     // 4718592
  const size_t R36T = (size_t)ROWS * R36;    // 1327104
  const size_t SCM  = (size_t)2 * BSZ * HALFC * NST * NCHUNK;  // 131072
  const size_t SCG  = (size_t)2 * BSZ * DI    * NST * NCHUNK;  // 262144
  float* Ym    = ws;                 // in-proj mixer, row-major   (later reused: convM)
  float* Yg    = ws + RD;            // in-proj global, row-major  (later reused: convG)
  float* xzT   = ws + 2 * RD;        // mixer pre-conv, ch-major   (later reused: yM)
  float* ghT   = ws + 3 * RD;        // global pre-conv, ch-major  (later reused: yG)
  float* xdblM = ws + 4 * RD;
  float* xdblG = ws + 4 * RD + R36T;
  float* dtM   = ws + 4 * RD + 2 * R36T;                    // [2*BSZ,64,L]
  float* dtG   = dtM + (size_t)2 * BSZ * HALFC * LSEQ;      // [2*BSZ,128,L]
  float* HendM = dtG + (size_t)2 * BSZ * DI * LSEQ;
  float* PprM  = HendM + SCM;
  float* HinM  = PprM + SCM;
  float* HendG = HinM + SCM;
  float* PprG  = HendG + SCG;
  float* HinG  = PprG + SCG;
  float* convM = Ym;   // safe reuse: Ym consumed by ln_transpose before conv writes
  float* convG = Yg;
  float* yM    = xzT;  // safe reuse: xzT consumed by conv before scan writes
  float* yG    = ghT;

  // 1) in-projections (WMMA)
  inproj_gemm<<<dim3(ROWS / 16, 16), 32, 0, stream>>>(f1, f2, Wi, bi, Wg, bg, Ym, Yg);
  // 2) layernorm + transpose to channel-major
  ln_transpose<<<ROWS, DI, 0, stream>>>(Ym, Yg, ln_g, ln_b, xzT, ghT);
  // 3) depthwise conv + SiLU
  const int tot = 2 * BSZ * DI * LSEQ;                      // 4718592
  dwconv_silu<<<tot / 256, 256, 0, stream>>>(xzT, wcx, bcx, wcz, bcz, convM);
  dwconv_silu<<<tot / 256, 256, 0, stream>>>(ghT, wcg, bcg, wcg + HALFC * 4, bcg + HALFC, convG);
  // 4) x_dbl projections (r = 36)
  xdbl_proj<<<(ROWS * R36) / 256, 256, 0, stream>>>(convM, Wxp_m, HALFC, xdblM);
  xdbl_proj<<<(ROWS * R36) / 256, 256, 0, stream>>>(convG, Wxp_g, DI, xdblG);
  // 5) dt projections (rank 4)
  dt_proj<<<(2 * BSZ * HALFC * LSEQ) / 256, 256, 0, stream>>>(xdblM, Wdt_m, HALFC, dtM);
  dt_proj<<<(2 * BSZ * DI * LSEQ) / 256, 256, 0, stream>>>(xdblG, Wdt_g, DI, dtG);
  // 6) chunked selective scans: phase1 (local) -> combine (carries) -> phase2 (emit y)
  const int wavesM = 2 * BSZ * (HALFC / 2);   // 128
  const int wavesG = 2 * BSZ * (DI / 2);      // 256
  scan_phase1<<<dim3(wavesM, NCHUNK), 32, 0, stream>>>(dtM, convM, xdblM, Alog_m, bdt_m, HALFC, HendM, PprM);
  scan_phase1<<<dim3(wavesG, NCHUNK), 32, 0, stream>>>(dtG, convG, xdblG, Alog_g, bdt_g, DI, HendG, PprG);
  scan_combine<<<wavesM, 32, 0, stream>>>(HendM, PprM, HALFC, HinM);
  scan_combine<<<wavesG, 32, 0, stream>>>(HendG, PprG, DI, HinG);
  scan_phase2<<<dim3(wavesM, NCHUNK), 32, 0, stream>>>(dtM, convM, xdblM, Alog_m, Dm, bdt_m, HinM, HALFC, yM);
  scan_phase2<<<dim3(wavesG, NCHUNK), 32, 0, stream>>>(dtG, convG, xdblG, Alog_g, Dg, bdt_g, HinG, DI, yG);
  // 7) fused Hadamard + output GEMM (WMMA), stores o1 then o2
  outproj_gemm<<<dim3((BSZ * LSEQ) / 16, 4, 2), 32, 0, stream>>>(yM, convM, yG, Wo, bo, out);
}
